// TMB_85813446574737
// MI455X (gfx1250) — matrix operand order repaired and checked by reading, not compiled, not run on hardware
//
#include <hip/hip_runtime.h>
#include <hip/hip_bf16.h>
#include <math.h>

// CDNA5 / gfx1250, wave32. All matmuls via v_wmma_f32_16x16x32_f16 (f32 accum).
// 32-bit index arithmetic everywhere (all tensors < 2^31 elements).

typedef _Float16 v16h __attribute__((ext_vector_type(16)));
typedef float    v8f  __attribute__((ext_vector_type(8)));
typedef int      v4i  __attribute__((ext_vector_type(4)));

union V16H {
    v16h     v;
    _Float16 h[16];
    float4   f4[2];
};

union H2 {
    unsigned u;
    _Float16 h[2];
};

// ---- CDNA5 async global->LDS copy (ASYNCcnt path), with safe fallback ------
#if defined(__has_builtin)
#if __has_builtin(__builtin_amdgcn_global_load_async_to_lds_b128)
#define HAVE_ASYNC_LDS 1
#endif
#endif
#ifndef HAVE_ASYNC_LDS
#define HAVE_ASYNC_LDS 0
#endif

#if defined(__has_builtin)
#if __has_builtin(__builtin_amdgcn_s_wait_asynccnt)
#define WAIT_ASYNCCNT0() __builtin_amdgcn_s_wait_asynccnt(0)
#endif
#endif
#ifndef WAIT_ASYNCCNT0
#define WAIT_ASYNCCNT0() asm volatile("s_wait_asynccnt 0" ::: "memory")
#endif

__device__ __forceinline__ float wave_sum32(float x) {
#pragma unroll
    for (int m = 16; m >= 1; m >>= 1) x += __shfl_xor(x, m, 32);
    return x;
}

// one v_exp_f32 + one v_rcp_f32
__device__ __forceinline__ float sigmoidf(float x) {
    return __builtin_amdgcn_rcpf(1.0f + __expf(-x));
}

// ---------------------------------------------------------------------------
// Kernel 0: pre-swizzle 64x64 f32 weights into WMMA B-operand order (f16).
// B operand (16x16x32 f16): lane L holds B[k(t)][n], t=0..15, n=L&15,
// k(t) = ks*32 + (L>>4)*8 + (t&7) + (t>>3)*16  -> one contiguous 32B load per lane per k-step.
// ---------------------------------------------------------------------------
__global__ __launch_bounds__(256)
void k_wconv(const float* __restrict__ Wp, const float* __restrict__ Wg,
             const float* __restrict__ Wog, const float* __restrict__ Wz,
             _Float16* __restrict__ Wsw)
{
    const float* Ws[4] = {Wp, Wg, Wog, Wz};
#pragma unroll
    for (int mat = 0; mat < 4; ++mat) {
        const float* W = Ws[mat];
        for (unsigned idx = threadIdx.x; idx < 4096u; idx += 256u) {
            unsigned t  = idx & 15u;
            unsigned L  = (idx >> 4) & 31u;
            unsigned ks = (idx >> 9) & 1u;
            unsigned nt = idx >> 10;
            unsigned row = ks * 32u + (L >> 4) * 8u + (t & 7u) + (t >> 3) * 16u;   // K
            unsigned col = nt * 16u + (L & 15u);            // N
            Wsw[mat * 4096u + idx] = (_Float16)W[row * 64u + col];
        }
    }
}

// ---------------------------------------------------------------------------
// Kernel 1: fused LN(LN(E)) -> x (f16, LDS); x@{Wp,Wg,Wog} via WMMA.
// 32 rows of [B*N*N,64] per 256-thread workgroup.
//   a_t  [B,H,N,N] f16 : transposed scatter (paired 32-bit stores)
//   gout [B*N*N,64] f16 : sigmoid(x@Wog+bog)
// ---------------------------------------------------------------------------
__global__ __launch_bounds__(256)
void k_proj(const float* __restrict__ E, const float* __restrict__ mask,
            const float* __restrict__ ln1_g, const float* __restrict__ ln1_b,
            const float* __restrict__ lnin_g, const float* __restrict__ lnin_b,
            const float* __restrict__ bp, const float* __restrict__ bg,
            const float* __restrict__ bog,
            const _Float16* __restrict__ Wsw,
            _Float16* __restrict__ a_t, _Float16* __restrict__ gout)
{
    __shared__ __attribute__((aligned(16))) _Float16 xs[32][72]; // 144B row stride

    const unsigned lane = threadIdx.x & 31u;
    const unsigned wv   = threadIdx.x >> 5;
    const unsigned rowbase = blockIdx.x * 32u;

    for (unsigned r = 0; r < 4; ++r) {
        const unsigned row = wv * 4u + r;
        const unsigned gr  = rowbase + row;
        const float* e = E + gr * 64u;
        float x0 = e[lane], x1 = e[lane + 32u];
        float mu = wave_sum32(x0 + x1) * (1.0f / 64.0f);
        float d0 = x0 - mu, d1 = x1 - mu;
        float var = wave_sum32(d0 * d0 + d1 * d1) * (1.0f / 64.0f);
        float rs = rsqrtf(var + 1e-5f);
        float y0 = d0 * rs * ln1_g[lane]       + ln1_b[lane];
        float y1 = d1 * rs * ln1_g[lane + 32u] + ln1_b[lane + 32u];
        mu = wave_sum32(y0 + y1) * (1.0f / 64.0f);
        d0 = y0 - mu; d1 = y1 - mu;
        var = wave_sum32(d0 * d0 + d1 * d1) * (1.0f / 64.0f);
        rs = rsqrtf(var + 1e-5f);
        xs[row][lane]       = (_Float16)(d0 * rs * lnin_g[lane]       + lnin_b[lane]);
        xs[row][lane + 32u] = (_Float16)(d1 * rs * lnin_g[lane + 32u] + lnin_b[lane + 32u]);
    }
    __syncthreads();

    const unsigned m16 = lane & 15u;
    const unsigned sel = lane >> 4;

    for (unsigned job = wv; job < 16u; job += 8u) {
        const unsigned tile = job >> 3;
        const unsigned sub  = job & 7u;
        const unsigned nt   = sub & 3u;
        const bool     isO  = (sub >= 4u);

        V16H A[2];
#pragma unroll
        for (unsigned ks = 0; ks < 2; ++ks) {
            const unsigned kb = ks * 32u + sel * 8u;
            const _Float16* src = &xs[tile * 16u + m16][0];
            A[ks].f4[0] = *(const float4*)(src + kb);
            A[ks].f4[1] = *(const float4*)(src + kb + 16u);
        }

        const unsigned h = nt * 16u + m16;

        if (!isO) {
            v8f pacc, gacc;
            const float bpv = bp[h], bgv = bg[h];
#pragma unroll
            for (int v = 0; v < 8; ++v) { pacc[v] = bpv; gacc[v] = bgv; }
#pragma unroll
            for (unsigned ks = 0; ks < 2; ++ks) {
                V16H B0, B1;
                const _Float16* p0 = Wsw + ((nt * 2u + ks) * 32u + lane) * 16u;
                B0.f4[0] = ((const float4*)p0)[0];
                B0.f4[1] = ((const float4*)p0)[1];
                pacc = __builtin_amdgcn_wmma_f32_16x16x32_f16(
                    false, A[ks].v, false, B0.v, (short)0, pacc, false, false);
                const _Float16* p1 = p0 + 4096u;
                B1.f4[0] = ((const float4*)p1)[0];
                B1.f4[1] = ((const float4*)p1)[1];
                gacc = __builtin_amdgcn_wmma_f32_16x16x32_f16(
                    false, A[ks].v, false, B1.v, (short)0, gacc, false, false);
            }
#pragma unroll
            for (int v = 0; v < 8; v += 2) {
                const unsigned M  = (unsigned)v + sel * 8u;
                const unsigned gr = rowbase + tile * 16u + M;
                const unsigned k  = gr & 255u;
                const unsigned i  = (gr >> 8) & 255u;
                const unsigned b  = gr >> 16;
                H2 pk;
                pk.h[0] = (_Float16)(mask[gr]      * sigmoidf(gacc[v])     * pacc[v]);
                pk.h[1] = (_Float16)(mask[gr + 1u] * sigmoidf(gacc[v + 1]) * pacc[v + 1]);
                *(unsigned*)(a_t + (((b * 64u + h) << 16) + (i << 8) + k)) = pk.u;
            }
        } else {
            v8f oacc;
            const float bov = bog[h];
#pragma unroll
            for (int v = 0; v < 8; ++v) oacc[v] = bov;
#pragma unroll
            for (unsigned ks = 0; ks < 2; ++ks) {
                V16H B2;
                const _Float16* p2 = Wsw + 2u * 4096u + ((nt * 2u + ks) * 32u + lane) * 16u;
                B2.f4[0] = ((const float4*)p2)[0];
                B2.f4[1] = ((const float4*)p2)[1];
                oacc = __builtin_amdgcn_wmma_f32_16x16x32_f16(
                    false, A[ks].v, false, B2.v, (short)0, oacc, false, false);
            }
#pragma unroll
            for (int v = 0; v < 8; ++v) {
                const unsigned M  = (unsigned)v + sel * 8u;
                const unsigned gr = rowbase + tile * 16u + M;
                gout[gr * 64u + h] = (_Float16)sigmoidf(oacc[v]);
            }
        }
    }
}

// ---------------------------------------------------------------------------
// Kernel 2: triangle einsum, LDS-resident. One workgroup per (b,h); the whole
// 256x256 f16 slice (128KB, padded rows 528B -> 132KB) is staged into LDS via
// async global->LDS (ASYNCcnt) and reused by all 136 symmetric tile pairs.
// Off-diagonal tiles are mirror-stored; two accumulators break the WMMA chain.
// ---------------------------------------------------------------------------
#define TRI_ROW_H 264u              // halves per padded LDS row (528 B)
#define TRI_LDS_BYTES (256u * 528u) // 135168 B

__global__ __launch_bounds__(256)
void k_tri(const _Float16* __restrict__ a_t, _Float16* __restrict__ t_ws)
{
    extern __shared__ _Float16 smem[];   // 256 rows x 264 halves

    const unsigned lane = threadIdx.x & 31u;
    const unsigned wv   = threadIdx.x >> 5;
    const unsigned h    = blockIdx.x & 63u;
    const unsigned b    = blockIdx.x >> 6;

    // ---- stage the (b,h) slice: 8192 x 16B chunks ----
    const char* gbase = (const char*)(a_t + ((b * 64u + h) << 16));
    char* lbase = (char*)smem;
    for (unsigned c = threadIdx.x; c < 8192u; c += 256u) {
        const unsigned row  = c >> 5;
        const unsigned col  = c & 31u;
        const unsigned loff = row * 528u + col * 16u;
#if HAVE_ASYNC_LDS
        __builtin_amdgcn_global_load_async_to_lds_b128(
            (__attribute__((address_space(1))) v4i*)(gbase + c * 16u),
            (__attribute__((address_space(3))) v4i*)(lbase + loff), 0, 0);
#else
        *(float4*)(lbase + loff) = *(const float4*)(gbase + c * 16u);
#endif
    }
#if HAVE_ASYNC_LDS
    WAIT_ASYNCCNT0();
#endif
    __syncthreads();

    const unsigned m16 = lane & 15u;
    const unsigned sel = lane >> 4;

    // ---- 136 tile pairs (it <= jt), 17 per wave ----
    for (unsigned p = wv; p < 136u; p += 8u) {
        int jt = (int)((sqrtf(8.0f * (float)p + 1.0f) - 1.0f) * 0.5f);
        while (((jt + 1) * (jt + 2)) / 2 <= (int)p) ++jt;
        while ((jt * (jt + 1)) / 2 > (int)p) --jt;
        const unsigned it = p - (unsigned)((jt * (jt + 1)) / 2);

        const _Float16* arow = smem + (it * 16u + m16) * TRI_ROW_H;
        const _Float16* brow = smem + ((unsigned)jt * 16u + m16) * TRI_ROW_H;

        v8f acc0 = {0.f,0.f,0.f,0.f,0.f,0.f,0.f,0.f};
        v8f acc1 = {0.f,0.f,0.f,0.f,0.f,0.f,0.f,0.f};
#pragma unroll
        for (unsigned ks = 0; ks < 8; ks += 2) {
            V16H A0, B0, A1, B1;
            const unsigned kb = ks * 32u + sel * 8u;
            const unsigned ko = ks * 32u + sel * 8u;
            A0.f4[0] = *(const float4*)(arow + kb);
            A0.f4[1] = *(const float4*)(arow + kb + 16u);
            B0.f4[0] = *(const float4*)(brow + ko);
            B0.f4[1] = *(const float4*)(brow + ko + 16u);
            A1.f4[0] = *(const float4*)(arow + kb + 32u);
            A1.f4[1] = *(const float4*)(arow + kb + 48u);
            B1.f4[0] = *(const float4*)(brow + ko + 32u);
            B1.f4[1] = *(const float4*)(brow + ko + 48u);
            acc0 = __builtin_amdgcn_wmma_f32_16x16x32_f16(
                false, A0.v, false, B0.v, (short)0, acc0, false, false);
            acc1 = __builtin_amdgcn_wmma_f32_16x16x32_f16(
                false, A1.v, false, B1.v, (short)0, acc1, false, false);
        }
        acc0 = acc0 + acc1;

        const unsigned i0 = it * 16u, j0 = (unsigned)jt * 16u;
#pragma unroll
        for (int v = 0; v < 8; ++v) {
            const unsigned M = (unsigned)v + sel * 8u;
            t_ws[((b * 256u + i0 + M) * 256u + j0 + m16) * 64u + h] = (_Float16)acc0[v];
        }
        if (it != (unsigned)jt) {   // wave-uniform branch
#pragma unroll
            for (int v = 0; v < 8; ++v) {
                const unsigned M = (unsigned)v + sel * 8u;
                t_ws[((b * 256u + j0 + m16) * 256u + i0 + M) * 64u + h] = (_Float16)acc0[v];
            }
        }
    }
}

// ---------------------------------------------------------------------------
// Kernel 3: out = E + g_out * (LN_H(t) @ Wz + bz).
// ---------------------------------------------------------------------------
__global__ __launch_bounds__(256)
void k_out(const float* __restrict__ E, const _Float16* __restrict__ t_ws,
           const float* __restrict__ lnout_g, const float* __restrict__ lnout_b,
           const float* __restrict__ bz,
           const _Float16* __restrict__ Wsw, const _Float16* __restrict__ gout,
           float* __restrict__ out)
{
    __shared__ __attribute__((aligned(16))) _Float16 us[32][72];

    const unsigned lane = threadIdx.x & 31u;
    const unsigned wv   = threadIdx.x >> 5;
    const unsigned rowbase = blockIdx.x * 32u;

    for (unsigned r = 0; r < 4; ++r) {
        const unsigned row = wv * 4u + r;
        const unsigned gr  = rowbase + row;
        const _Float16* t = t_ws + gr * 64u;
        float x0 = (float)t[lane], x1 = (float)t[lane + 32u];
        float mu = wave_sum32(x0 + x1) * (1.0f / 64.0f);
        float d0 = x0 - mu, d1 = x1 - mu;
        float var = wave_sum32(d0 * d0 + d1 * d1) * (1.0f / 64.0f);
        float rs = rsqrtf(var + 1e-5f);
        us[row][lane]       = (_Float16)(d0 * rs * lnout_g[lane]       + lnout_b[lane]);
        us[row][lane + 32u] = (_Float16)(d1 * rs * lnout_g[lane + 32u] + lnout_b[lane + 32u]);
    }
    __syncthreads();

    const unsigned m16  = lane & 15u;
    const unsigned sel  = lane >> 4;
    const unsigned tile = wv >> 2;
    const unsigned nt   = wv & 3u;
    const unsigned d    = nt * 16u + m16;

    v8f acc;
    const float bzv = bz[d];
#pragma unroll
    for (int v = 0; v < 8; ++v) acc[v] = bzv;

#pragma unroll
    for (unsigned ks = 0; ks < 2; ++ks) {
        V16H A, B;
        const unsigned kb = ks * 32u + sel * 8u;
        const _Float16* src = &us[tile * 16u + m16][0];
        A.f4[0] = *(const float4*)(src + kb);
        A.f4[1] = *(const float4*)(src + kb + 16u);
        const _Float16* pw = Wsw + 3u * 4096u + ((nt * 2u + ks) * 32u + lane) * 16u;
        B.f4[0] = ((const float4*)pw)[0];
        B.f4[1] = ((const float4*)pw)[1];
        acc = __builtin_amdgcn_wmma_f32_16x16x32_f16(
            false, A.v, false, B.v, (short)0, acc, false, false);
    }

#pragma unroll
    for (int v = 0; v < 8; ++v) {
        const unsigned M   = (unsigned)v + sel * 8u;
        const unsigned gr  = rowbase + tile * 16u + M;
        const unsigned idx = gr * 64u + d;
        out[idx] = E[idx] + (float)gout[idx] * acc[v];
    }
}

// ---------------------------------------------------------------------------
// Host launcher
// Inputs: 0:E 1:mask 2:ln1_g 3:ln1_b 4:lnin_g 5:lnin_b 6:Wp 7:bp 8:Wg 9:bg
//         10:lnout_g 11:lnout_b 12:Wog 13:bog 14:Wz 15:bz
// Workspace: Wsw (64KB slot) | a_t 64MB f16 | gout 64MB f16 | t 64MB f16
// ---------------------------------------------------------------------------
extern "C" void kernel_launch(void* const* d_in, const int* in_sizes, int n_in,
                              void* d_out, int out_size, void* d_ws, size_t ws_size,
                              hipStream_t stream) {
    (void)in_sizes; (void)n_in; (void)out_size; (void)ws_size;
    const float* E       = (const float*)d_in[0];
    const float* mask    = (const float*)d_in[1];
    const float* ln1_g   = (const float*)d_in[2];
    const float* ln1_b   = (const float*)d_in[3];
    const float* lnin_g  = (const float*)d_in[4];
    const float* lnin_b  = (const float*)d_in[5];
    const float* Wp      = (const float*)d_in[6];
    const float* bp      = (const float*)d_in[7];
    const float* Wg      = (const float*)d_in[8];
    const float* bg      = (const float*)d_in[9];
    const float* lnout_g = (const float*)d_in[10];
    const float* lnout_b = (const float*)d_in[11];
    const float* Wog     = (const float*)d_in[12];
    const float* bog     = (const float*)d_in[13];
    const float* Wz      = (const float*)d_in[14];
    const float* bz      = (const float*)d_in[15];
    float* outp = (float*)d_out;

    char* wsb = (char*)d_ws;
    const size_t SLOT = 67108864; // 64 MiB per f16 tensor [B,N,N,64]
    _Float16* Wsw = (_Float16*)(wsb);
    _Float16* a_t = (_Float16*)(wsb + 65536);
    _Float16* go  = (_Float16*)(wsb + 65536 + SLOT);
    _Float16* tws = (_Float16*)(wsb + 65536 + 2 * SLOT);

    k_wconv<<<1, 256, 0, stream>>>(Wp, Wg, Wog, Wz, Wsw);

    // 524288 rows / 32 per block
    k_proj<<<16384, 256, 0, stream>>>(E, mask, ln1_g, ln1_b, lnin_g, lnin_b,
                                      bp, bg, bog, Wsw, a_t, go);

    // one workgroup per (b,h): 8*64 = 512 blocks, 132KB dynamic LDS each
    k_tri<<<512, 256, TRI_LDS_BYTES, stream>>>(a_t, tws);

    k_out<<<16384, 256, 0, stream>>>(E, tws, lnout_g, lnout_b, bz, Wsw, go, outp);
}
